// Attention_60696477827159
// MI455X (gfx1250) — compile-verified, matrix operand order checked by reading, not run on hardware
//
#include <hip/hip_runtime.h>
#include <hip/hip_bf16.h>

#define BB   64
#define NN   512
#define DD   1024
#define HH   16
#define DPH  64
#define EPS  1e-5f
#define QSCALE 0.125f   // 1/sqrt(64)

typedef __attribute__((ext_vector_type(16))) __bf16    v16bf;
typedef __attribute__((ext_vector_type(8)))  float     v8f;
typedef __attribute__((ext_vector_type(4)))  unsigned  v4u;
typedef __attribute__((ext_vector_type(8)))  unsigned  v8u;

#if __has_builtin(__builtin_amdgcn_s_wait_tensorcnt)
#define WAIT_TENSORCNT(n) __builtin_amdgcn_s_wait_tensorcnt(n)
#else
#define WAIT_TENSORCNT(n) asm volatile("s_wait_tensorcnt " #n ::: "memory")
#endif

static __device__ inline v8f vzero8() {
  v8f z;
#pragma unroll
  for (int i = 0; i < 8; ++i) z[i] = 0.0f;
  return z;
}

static __device__ inline unsigned pk2bf(float x, float y) {
  unsigned short ux = __builtin_bit_cast(unsigned short, (__bf16)x);
  unsigned short uy = __builtin_bit_cast(unsigned short, (__bf16)y);
  return (unsigned)ux | ((unsigned)uy << 16);
}

// TDM: load one 64x64 bf16 tile (row stride DD elems) from global into LDS.
// D# group0/group1 per CDNA5 ISA 8.3/8.4; 2D tensor -> VADDR2/3 omitted.
static __device__ inline void tdm_load_tile(unsigned lds_off, const __bf16* gptr) {
  unsigned long long ga = (unsigned long long)(uintptr_t)gptr;
  v4u g0;
  g0[0] = 1u;                                                // count=1, user desc
  g0[1] = lds_off;                                           // lds_addr (bytes)
  g0[2] = (unsigned)ga;                                      // global_addr[31:0]
  g0[3] = (unsigned)((ga >> 32) & 0x1FFFFFFu) | (2u << 30);  // addr[56:32] | type=2
  v8u g1;
  g1[0] = (1u << 16);                  // workgroup_mask=0, data_size=1 (2 bytes)
  g1[1] = ((unsigned)DD & 0xFFFFu) << 16;          // tensor_dim0[15:0] = 1024
  g1[2] = ((unsigned)DD >> 16) | (64u << 16);      // tensor_dim0[31:16] | tensor_dim1=64
  g1[3] = (64u << 16);                 // tensor_dim1 hi=0 | tile_dim0=64
  g1[4] = 64u;                         // tile_dim1=64, tile_dim2=0
  g1[5] = (unsigned)DD;                // tensor_dim0_stride[31:0] = 1024
  g1[6] = 0u;                          // stride hi / dim1_stride lo
  g1[7] = 0u;                          // dim1_stride hi
  asm volatile("tensor_load_to_lds %0, %1" :: "s"(g0), "s"(g1) : "memory");
}

// ---------------- kernel 1: convert Wk / Wv to bf16 into workspace ----------
__global__ __launch_bounds__(256)
void prep_weights(const float* __restrict__ Wk, const float* __restrict__ Wv,
                  __bf16* __restrict__ Wk_bf, __bf16* __restrict__ Wv_bf) {
  int i = blockIdx.x * blockDim.x + threadIdx.x;   // over (H*DPH*D)/4
  float4 a = ((const float4*)Wk)[i];
  float4 b = ((const float4*)Wv)[i];
  uint2* ko = (uint2*)(Wk_bf + 4 * (size_t)i);
  uint2* vo = (uint2*)(Wv_bf + 4 * (size_t)i);
  ko[0] = make_uint2(pk2bf(a.x, a.y), pk2bf(a.z, a.w));
  vo[0] = make_uint2(pk2bf(b.x, b.y), pk2bf(b.z, b.w));
}

// ---------------- kernel 2: q projection (tiny: 134 MFLOP) -----------------
__global__ __launch_bounds__(256)
void qproj(const float* __restrict__ queries, const float* __restrict__ Wq,
           const float* __restrict__ bq, float* __restrict__ q_out) {
  int id = blockIdx.x * blockDim.x + threadIdx.x;  // B*H*DPH = 65536
  int b  = id >> 10;          // / (H*DPH)
  int he = id & 1023;         // h*DPH + e
  const float4* qr = (const float4*)(queries + (size_t)b * DD);
  const float4* wr = (const float4*)(Wq + (size_t)he * DD);
  float acc = 0.0f;
#pragma unroll 4
  for (int i = 0; i < DD / 4; ++i) {
    float4 x = qr[i], w = wr[i];
    acc += x.x * w.x + x.y * w.y + x.z * w.z + x.w * w.w;
  }
  q_out[id] = (acc + bq[he]) * QSCALE;   // fold 1/sqrt(DPH) into q
}

// ---------------- kernel 3: fused K/V projection + attention ---------------
// grid = B*H workgroups; 256 threads = 8 waves; M=32 rows per wave.
// LDS carve (bytes):
//   sAk  256*64 bf16        = 32768
//   sAv  256*64 bf16        = 32768
//   sWk  2 x 64*64 bf16     = 16384   (TDM double buffer)
//   sWv  2 x 64*64 bf16     = 16384   (TDM double buffer)
//   sV   512*64 bf16        = 65536
//   sQ       64 f32         =   256
//   sS      512 f32         =  2048
//   sRed    256 f32         =  1024
//   sPart  4*64 f32         =  1024
#define SMEM_BYTES (32768 + 32768 + 16384 + 16384 + 65536 + 256 + 2048 + 1024 + 1024)

__global__ __launch_bounds__(256)
void attn_fused(const float* __restrict__ keys, const float* __restrict__ values,
                const __bf16* __restrict__ Wk_bf, const __bf16* __restrict__ Wv_bf,
                const float* __restrict__ bk, const float* __restrict__ bv,
                const float* __restrict__ q_proj, float* __restrict__ attn_buf) {
  extern __shared__ char smem_raw[];
  __bf16* sAk  = (__bf16*)smem_raw;         // [256][64]
  __bf16* sAv  = sAk + 256 * 64;            // [256][64]
  __bf16* sWk  = sAv + 256 * 64;            // [2][64][64]
  __bf16* sWv  = sWk + 2 * 64 * 64;         // [2][64][64]
  __bf16* sV   = sWv + 2 * 64 * 64;         // [512][64]
  float*  sQ   = (float*)(sV + 512 * 64);   // [64]
  float*  sS   = sQ + 64;                   // [512]
  float*  sRed = sS + 512;                  // [256]
  float*  sPart= sRed + 256;                // [4][64]

  const int b = blockIdx.x / HH;
  const int h = blockIdx.x % HH;
  const int tid  = threadIdx.x;
  const int wave = tid >> 5;
  const int lane = tid & 31;
  const int l16  = lane & 15;
  const int hi   = lane >> 4;

  // q vector (f32) copied global->LDS on the ASYNC path: 16 lanes x 16 B.
  if (tid < 16) {
    const float* gq = q_proj + ((size_t)b * HH + h) * DPH + tid * 4;
    unsigned lq = (unsigned)(uintptr_t)(sQ + tid * 4);
    asm volatile("global_load_async_to_lds_b128 %0, %1, off"
                 :: "v"(lq), "v"(gq) : "memory");
  }
  asm volatile("s_wait_asynccnt 0x0" ::: "memory");

  const __bf16* wkh = Wk_bf + (size_t)h * DPH * DD;   // head's 64x1024 block
  const __bf16* wvh = Wv_bf + (size_t)h * DPH * DD;
  const unsigned sWkOff = (unsigned)(uintptr_t)sWk;   // LDS byte offsets
  const unsigned sWvOff = (unsigned)(uintptr_t)sWv;

  const size_t kvBase = (size_t)b * NN * DD;

  for (int t = 0; t < 2; ++t) {               // 2 key tiles of 256 rows
    const int n0 = t * 256;
    v8f accK[2][4], accV[2][4];
#pragma unroll
    for (int g = 0; g < 2; ++g)
#pragma unroll
      for (int j = 0; j < 4; ++j) { accK[g][j] = vzero8(); accV[g][j] = vzero8(); }

    for (int c = 0; c < 16; ++c) {            // D in chunks of 64
      const int d0 = c * 64;
      // ---- stage keys/values tile (f32 -> packed bf16, one row per thread) ----
      {
        const float4* ksrc = (const float4*)(keys   + kvBase + (size_t)(n0 + tid) * DD + d0);
        const float4* vsrc = (const float4*)(values + kvBase + (size_t)(n0 + tid) * DD + d0);
        uint4* kdst = (uint4*)(sAk + tid * 64);
        uint4* vdst = (uint4*)(sAv + tid * 64);
#pragma unroll
        for (int i = 0; i < 8; ++i) {
          float4 f0 = ksrc[2 * i], f1 = ksrc[2 * i + 1];
          kdst[i] = make_uint4(pk2bf(f0.x, f0.y), pk2bf(f0.z, f0.w),
                               pk2bf(f1.x, f1.y), pk2bf(f1.z, f1.w));
          float4 g0 = vsrc[2 * i], g1 = vsrc[2 * i + 1];
          vdst[i] = make_uint4(pk2bf(g0.x, g0.y), pk2bf(g0.z, g0.w),
                               pk2bf(g1.x, g1.y), pk2bf(g1.z, g1.w));
        }
        if (c + 1 < 16) {                     // gfx1250 global_prefetch_b8
          __builtin_prefetch(keys   + kvBase + (size_t)(n0 + tid) * DD + d0 + 64, 0, 1);
          __builtin_prefetch(values + kvBase + (size_t)(n0 + tid) * DD + d0 + 64, 0, 1);
        }
      }
      // ---- Wk/Wv chunks via Tensor Data Mover, double-buffered on TENSORcnt ----
      if (wave == 0) {
        if (c == 0) {                         // prime buffer 0 with chunk 0
          tdm_load_tile(sWkOff, wkh);
          tdm_load_tile(sWvOff, wvh);
        }
        if (c + 1 < 16) {                     // prefetch chunk c+1 into other buffer
          const unsigned nb = (unsigned)((c + 1) & 1) * 64 * 64 * 2;
          tdm_load_tile(sWkOff + nb, wkh + d0 + 64);
          tdm_load_tile(sWvOff + nb, wvh + d0 + 64);
          WAIT_TENSORCNT(2);                  // chunk c's two tiles complete
        } else {
          WAIT_TENSORCNT(0);
        }
      }
      __syncthreads();
      // ---- WMMA: 2 K-steps of 32; B-frags reused across 2 row-groups ----
      const __bf16* wkBuf = sWk + (c & 1) * 64 * 64;
      const __bf16* wvBuf = sWv + (c & 1) * 64 * 64;
#pragma unroll
      for (int kk = 0; kk < 2; ++kk) {
        const int koff = kk * 32 + hi * 16;
        v16bf aK[2], aV[2];
#pragma unroll
        for (int g = 0; g < 2; ++g) {
          const int row = wave * 32 + g * 16 + l16;
          aK[g] = *(const v16bf*)(sAk + row * 64 + koff);
          aV[g] = *(const v16bf*)(sAv + row * 64 + koff);
        }
#pragma unroll
        for (int j = 0; j < 4; ++j) {
          v16bf bK = *(const v16bf*)(wkBuf + (j * 16 + l16) * 64 + koff);
          v16bf bV = *(const v16bf*)(wvBuf + (j * 16 + l16) * 64 + koff);
#pragma unroll
          for (int g = 0; g < 2; ++g) {
            accK[g][j] = __builtin_amdgcn_wmma_f32_16x16x32_bf16(
                false, aK[g], false, bK, (short)0, accK[g][j], false, false);
            accV[g][j] = __builtin_amdgcn_wmma_f32_16x16x32_bf16(
                false, aV[g], false, bV, (short)0, accV[g][j], false, false);
          }
        }
      }
      __syncthreads();
    }

    // ---- biases (C/D layout: lane holds e = 16j+l16, rows r(+8 if hi)) ----
#pragma unroll
    for (int j = 0; j < 4; ++j) {
      const float bkv = bk[h * DPH + j * 16 + l16];
      const float bvv = bv[h * DPH + j * 16 + l16];
#pragma unroll
      for (int g = 0; g < 2; ++g)
#pragma unroll
        for (int r = 0; r < 8; ++r) { accK[g][j][r] += bkv; accV[g][j][r] += bvv; }
    }

    // ---- scores: w[m] = sum_e q[e]*k[m][e]; reduce across 16-lane groups ----
    float qv[4];
#pragma unroll
    for (int j = 0; j < 4; ++j) qv[j] = sQ[j * 16 + l16];
#pragma unroll
    for (int g = 0; g < 2; ++g)
#pragma unroll
      for (int r = 0; r < 8; ++r) {
        float p = 0.0f;
#pragma unroll
        for (int j = 0; j < 4; ++j) p += qv[j] * accK[g][j][r];
        p += __shfl_xor(p, 1, 32);
        p += __shfl_xor(p, 2, 32);
        p += __shfl_xor(p, 4, 32);
        p += __shfl_xor(p, 8, 32);
        if (l16 == 0) sS[n0 + wave * 32 + g * 16 + r + hi * 8] = p;
      }

    // ---- park projected V tile in LDS as bf16 ----
#pragma unroll
    for (int g = 0; g < 2; ++g)
#pragma unroll
      for (int j = 0; j < 4; ++j)
#pragma unroll
        for (int r = 0; r < 8; ++r)
          sV[(size_t)(n0 + wave * 32 + g * 16 + r + hi * 8) * 64 + j * 16 + l16] =
              (__bf16)accV[g][j][r];
  }
  __syncthreads();

  // ---- softmax over sS[512] ----
  float lm = fmaxf(sS[tid], sS[tid + 256]);
  sRed[tid] = lm;
  __syncthreads();
  for (int s = 128; s > 0; s >>= 1) {
    if (tid < s) sRed[tid] = fmaxf(sRed[tid], sRed[tid + s]);
    __syncthreads();
  }
  const float mx = sRed[0];
  __syncthreads();
  float e0 = __expf(sS[tid] - mx);
  float e1 = __expf(sS[tid + 256] - mx);
  sS[tid] = e0; sS[tid + 256] = e1;
  sRed[tid] = e0 + e1;
  __syncthreads();
  for (int s = 128; s > 0; s >>= 1) {
    if (tid < s) sRed[tid] += sRed[tid + s];
    __syncthreads();
  }
  const float denom = sRed[0];

  // ---- attn[e] = sum_n p_n * v[n][e] / denom ----
  {
    const int e = tid & 63, part = tid >> 6;
    float acc = 0.0f;
    const int nstart = part * 128;
    for (int n = nstart; n < nstart + 128; ++n)
      acc += sS[n] * (float)sV[(size_t)n * 64 + e];
    sPart[part * 64 + e] = acc;
  }
  __syncthreads();
  if (tid < 64) {
    float a = (sPart[tid] + sPart[64 + tid] + sPart[128 + tid] + sPart[192 + tid]) / denom;
    attn_buf[(size_t)b * DD + h * DPH + tid] = a;
  }
}

// ---------------- kernel 4: residual + LayerNorm ---------------------------
__global__ __launch_bounds__(256)
void layernorm(const float* __restrict__ attn_buf, const float* __restrict__ queries,
               const float* __restrict__ gamma, const float* __restrict__ beta,
               float* __restrict__ out) {
  __shared__ float red[256];
  const int b = blockIdx.x, tid = threadIdx.x;
  float x[4];
  float s = 0.0f;
#pragma unroll
  for (int i = 0; i < 4; ++i) {
    const int idx = tid + i * 256;
    x[i] = attn_buf[(size_t)b * DD + idx] + queries[(size_t)b * DD + idx];
    s += x[i];
  }
  red[tid] = s; __syncthreads();
  for (int st = 128; st > 0; st >>= 1) { if (tid < st) red[tid] += red[tid + st]; __syncthreads(); }
  const float mu = red[0] * (1.0f / DD);
  __syncthreads();
  float vs = 0.0f;
#pragma unroll
  for (int i = 0; i < 4; ++i) { const float d = x[i] - mu; vs += d * d; }
  red[tid] = vs; __syncthreads();
  for (int st = 128; st > 0; st >>= 1) { if (tid < st) red[tid] += red[tid + st]; __syncthreads(); }
  const float rstd = rsqrtf(red[0] * (1.0f / DD) + EPS);
#pragma unroll
  for (int i = 0; i < 4; ++i) {
    const int idx = tid + i * 256;
    out[(size_t)b * DD + idx] = (x[i] - mu) * rstd * gamma[idx] + beta[idx];
  }
}

// ---------------------------------------------------------------------------
extern "C" void kernel_launch(void* const* d_in, const int* in_sizes, int n_in,
                              void* d_out, int out_size, void* d_ws, size_t ws_size,
                              hipStream_t stream) {
  const float* queries = (const float*)d_in[0];
  const float* keys    = (const float*)d_in[1];
  const float* values  = (const float*)d_in[2];
  const float* Wq      = (const float*)d_in[3];
  const float* bq      = (const float*)d_in[4];
  const float* Wk      = (const float*)d_in[5];
  const float* bk      = (const float*)d_in[6];
  const float* Wv      = (const float*)d_in[7];
  const float* bv      = (const float*)d_in[8];
  const float* gamma   = (const float*)d_in[9];
  const float* beta    = (const float*)d_in[10];
  float* out = (float*)d_out;

  char* ws = (char*)d_ws;
  float*  q_proj   = (float*)ws;                                   // 256 KB
  __bf16* Wk_bf    = (__bf16*)(ws + (256u << 10));                 // 2 MB
  __bf16* Wv_bf    = (__bf16*)(ws + (256u << 10) + (2u << 20));    // 2 MB
  float*  attn_buf = (float*)(ws + (256u << 10) + (4u << 20));     // 256 KB

  prep_weights<<<(HH * DPH * DD / 4) / 256, 256, 0, stream>>>(Wk, Wv, Wk_bf, Wv_bf);
  qproj<<<(BB * HH * DPH) / 256, 256, 0, stream>>>(queries, Wq, bq, q_proj);
  attn_fused<<<BB * HH, 256, SMEM_BYTES, stream>>>(keys, values, Wk_bf, Wv_bf,
                                                   bk, bv, q_proj, attn_buf);
  layernorm<<<BB, 256, 0, stream>>>(attn_buf, queries, gamma, beta, out);
}
